// PSP_34522947125517
// MI455X (gfx1250) — matrix operand order repaired and checked by reading, not compiled, not run on hardware
//
#include <hip/hip_runtime.h>
#include <hip/hip_bf16.h>
#include <math.h>

// PSP (positive sample propagation) fused pipeline for gfx1250 (MI455X).
// All GEMM stages use v_wmma_f32_16x16x32_f16 (wave32 WMMA), f32 accumulate.
// beta ([8,2048,2048] f32, 128MB) lives in d_ws and stays L2-resident (192MB L2).
// Row-structured GEMM waves: one wave owns 16 rows x 128 cols (8 WMMA n-tiles):
// 64 acc VGPRs + ~64 hoisted B-tile VGPRs stays under the 256-VGPR ceiling
// (16 n-tiles/wave spilled accumulators to scratch in the previous build).
// Workspace requirement: ~161 MB.

typedef __attribute__((ext_vector_type(16))) _Float16 v16h;
typedef __attribute__((ext_vector_type(8)))  _Float16 v8h;
typedef __attribute__((ext_vector_type(8)))  float    v8f;

#define BB 8
#define TT 2048
#define DD 256
#define EPS 1e-8f

__device__ __forceinline__ v8f wmma16(v16h a, v16h b, v8f c) {
  return __builtin_amdgcn_wmma_f32_16x16x32_f16(
      /*neg_a=*/false, a, /*neg_b=*/false, b,
      /*c_mod=*/(short)0, c, /*reuse_a=*/false, /*reuse_b=*/false);
}

// A-operand (contiguous k order, not the ISA table order; B matches): lane holds row l%16; K chunk
// [base..base+15], base = k0 + (lane>=16 ? 16 : 0).
__device__ __forceinline__ v16h loadA_f32(const float* p0) {
  v16h r;
#pragma unroll
  for (int i = 0; i < 8; ++i) {
    r[i]     = (_Float16)p0[i];
    r[8 + i] = (_Float16)p0[8 + i];
  }
  return r;
}
__device__ __forceinline__ v16h loadA_h(const _Float16* p0) {
  v8h c0 = *(const v8h*)p0;
  v8h c1 = *(const v8h*)(p0 + 8);
  v16h r;
#pragma unroll
  for (int i = 0; i < 8; ++i) { r[i] = c0[i]; r[8 + i] = c1[i]; }
  return r;
}

// ---------------------------------------------------------------------------
// 0) f32 -> f16 conversion (weights, once)
__global__ void psp_cvt_f16(const float* __restrict__ src, _Float16* __restrict__ dst, int n) {
  const int i = blockIdx.x * 256 + threadIdx.x;
  if (i < n) dst[i] = (_Float16)src[i];
}

// ---------------------------------------------------------------------------
// 1) Projection: out = relu(X @ Wh^T), X:[16384,256] f32, Wh:[256,256] f16.
//    One wave: 16 rows x 128 cols (half). Optionally store normal f16 [BT,256]
//    and/or transposed f16 [B,256,2048].
__global__ void psp_proj_row(const float* __restrict__ X, const _Float16* __restrict__ Wh,
                             _Float16* __restrict__ outN, _Float16* __restrict__ outT) {
  const int tid = threadIdx.x, wid = tid >> 5, lane = tid & 31;
  const int l16 = lane & 15, hi = lane >> 4;
  const int w = blockIdx.x * 8 + wid;             // 2048 waves
  const int c0 = (w & 1) * 128;                   // column half
  const int t0 = (w >> 1) * 16;                   // 1024 row tiles over B*T
  const float* arow = X + (size_t)(t0 + l16) * DD;
  v8f acc[8] = {};
#pragma unroll
  for (int kk = 0; kk < 8; ++kk) {
    const int k0 = kk * 32;
    v16h a = loadA_f32(arow + k0 + (hi ? 16 : 0));
#pragma unroll
    for (int j = 0; j < 8; ++j) {
      v16h bv = *(const v16h*)(Wh + (size_t)(c0 + j * 16 + l16) * DD + k0 + (hi ? 16 : 0));
      acc[j] = wmma16(a, bv, acc[j]);
    }
  }
  if (outN) {
#pragma unroll
    for (int j = 0; j < 8; ++j)
#pragma unroll
      for (int r = 0; r < 8; ++r) {
        float v = acc[j][r]; v = v > 0.f ? v : 0.f;
        const int m = r + (hi ? 8 : 0);
        outN[(size_t)(t0 + m) * DD + c0 + j * 16 + l16] = (_Float16)v;
      }
  }
  if (outT) {
    const int b_ = t0 >> 11;                      // batch
    const int tin = (t0 & (TT - 1)) + (hi ? 8 : 0);
#pragma unroll
    for (int j = 0; j < 8; ++j) {
      v8h pk;
#pragma unroll
      for (int r = 0; r < 8; ++r) {
        float v = acc[j][r]; pk[r] = (_Float16)(v > 0.f ? v : 0.f);
      }
      *(v8h*)(outT + ((size_t)b_ * DD + c0 + j * 16 + l16) * TT + tin) = pk;
    }
  }
}

// ---------------------------------------------------------------------------
// 2) beta[b,t,s] = relu( (v2 . a1) / 16 ); one wave: 16 t-rows x 128 s-cols.
__global__ void psp_beta_row(const _Float16* __restrict__ v2h,
                             const _Float16* __restrict__ a1h,
                             float* __restrict__ beta) {
  const int tid = threadIdx.x, wid = tid >> 5, lane = tid & 31;
  const int l16 = lane & 15, hi = lane >> 4;
  const int w = blockIdx.x * 8 + wid;             // 16384 waves
  const int s0 = (w & 15) * 128;
  const int t0 = ((w >> 4) & 127) * 16;
  const int b  = w >> 11;
  const _Float16* ar = v2h + ((size_t)b * TT + t0 + l16) * DD;
  const _Float16* br = a1h + ((size_t)b * TT + s0) * DD;
  v8f acc[8] = {};
#pragma unroll
  for (int kk = 0; kk < 8; ++kk) {
    const int k0 = kk * 32;
    v16h a = loadA_h(ar + k0 + (hi ? 16 : 0));
#pragma unroll
    for (int j = 0; j < 8; ++j) {
      v16h bv = *(const v16h*)(br + (size_t)(j * 16 + l16) * DD + k0 + (hi ? 16 : 0));
      acc[j] = wmma16(a, bv, acc[j]);
    }
  }
#pragma unroll
  for (int j = 0; j < 8; ++j)
#pragma unroll
    for (int r = 0; r < 8; ++r) {
      const int m = r + (hi ? 8 : 0);
      float v = acc[j][r] * 0.0625f;              // 1/sqrt(256)
      beta[((size_t)b * TT + t0 + m) * TT + s0 + j * 16 + l16] = v > 0.f ? v : 0.f;
    }
}

// ---------------------------------------------------------------------------
// 3) Row sums S_t (one wave per row)
__global__ void psp_rowsum(const float* __restrict__ beta, float* __restrict__ S) {
  const int lane = threadIdx.x & 31, wid = threadIdx.x >> 5;
  const int row = blockIdx.x * 8 + wid;           // 16384
  const float* p = beta + (size_t)row * TT;
  float acc = 0.f;
  for (int i = lane; i < TT; i += 32) acc += p[i];
#pragma unroll
  for (int m = 16; m >= 1; m >>= 1) acc += __shfl_xor(acc, m, 32);
  if (lane == 0) S[row] = acc;
}

// 4) masked row sums -> thrR, scaleR
__global__ void psp_rowmask(const float* __restrict__ beta, const float* __restrict__ S,
                            const float* __restrict__ thrp,
                            float* __restrict__ thrR, float* __restrict__ scaleR) {
  const float thr = thrp[0] * (10.0f / (float)TT);
  const int lane = threadIdx.x & 31, wid = threadIdx.x >> 5;
  const int row = blockIdx.x * 8 + wid;
  const float Sp = S[row] + EPS;
  const float tr = thr * Sp;
  const float* p = beta + (size_t)row * TT;
  float acc = 0.f;
  for (int i = lane; i < TT; i += 32) { float v = p[i]; if (v > tr) acc += v; }
#pragma unroll
  for (int m = 16; m >= 1; m >>= 1) acc += __shfl_xor(acc, m, 32);
  if (lane == 0) {
    thrR[row] = tr;
    scaleR[row] = 1.0f / (Sp * (acc / Sp + EPS));
  }
}

// 5) Column sums C_s (coalesced: 256 consecutive columns per block)
__global__ void psp_colsum(const float* __restrict__ beta, float* __restrict__ C) {
  const int idx = blockIdx.x * 256 + threadIdx.x; // 16384
  const int b = idx >> 11, s = idx & (TT - 1);
  const float* p = beta + (size_t)b * TT * TT + s;
  float acc = 0.f;
  for (int t = 0; t < TT; ++t) acc += p[(size_t)t * TT];
  C[idx] = acc;
}

// 6) masked column sums -> thrC, scaleC; also pred (batch 0 diagonal)
__global__ void psp_colmask(const float* __restrict__ beta, const float* __restrict__ C,
                            const float* __restrict__ thrp,
                            float* __restrict__ thrC, float* __restrict__ scaleC,
                            float* __restrict__ pred) {
  const float thr = thrp[0] * (10.0f / (float)TT);
  const int idx = blockIdx.x * 256 + threadIdx.x;
  const int b = idx >> 11, s = idx & (TT - 1);
  const float Cp = C[idx] + EPS;
  const float tc = thr * Cp;
  const float* p = beta + (size_t)b * TT * TT + s;
  float acc = 0.f;
  for (int t = 0; t < TT; ++t) { float v = p[(size_t)t * TT]; if (v > tc) acc += v; }
  thrC[idx] = tc;
  scaleC[idx] = 1.0f / (Cp * (acc / Cp + EPS));
  if (b == 0) pred[s] = (p[(size_t)s * TT] > tc) ? 1.0f : 0.0f;
}

// ---------------------------------------------------------------------------
// 7) pspv = v_fea + g_va @ a2; one wave: 16 t-rows x 128 d-cols.
__global__ void psp_attn_va(const float* __restrict__ beta, const float* __restrict__ thrR,
                            const float* __restrict__ scaleR, const _Float16* __restrict__ a2T,
                            const float* __restrict__ v_fea, _Float16* __restrict__ pspv) {
  const int tid = threadIdx.x, wid = tid >> 5, lane = tid & 31;
  const int l16 = lane & 15, hi = lane >> 4;
  const int w = blockIdx.x * 8 + wid;             // 2048 waves
  const int c0 = (w & 1) * 128;                   // d column half
  const int t0 = ((w >> 1) & 127) * 16;
  const int b  = w >> 8;
  const int trow = b * TT + t0 + l16;
  const float tr = thrR[trow], sc = scaleR[trow];
  const float* brow = beta + (size_t)trow * TT;
  const _Float16* bTb = a2T + ((size_t)b * DD + c0) * TT;
  v8f acc[8] = {};
  for (int si = 0; si < 64; ++si) {
    const int s0 = si * 32;
    const float* p0 = brow + s0 + (hi ? 16 : 0);
    __builtin_prefetch(brow + s0 + 128, 0, 3);    // global_prefetch_b8: 4 chunks ahead
    v16h a;
#pragma unroll
    for (int j = 0; j < 8; ++j) {
      float v0 = p0[j], v1 = p0[8 + j];
      a[j]     = (_Float16)(v0 > tr ? v0 * sc : 0.f);
      a[8 + j] = (_Float16)(v1 > tr ? v1 * sc : 0.f);
    }
#pragma unroll
    for (int j = 0; j < 8; ++j) {
      v16h bv = *(const v16h*)(bTb + (size_t)(j * 16 + l16) * TT + s0 + (hi ? 16 : 0));
      acc[j] = wmma16(a, bv, acc[j]);
    }
  }
#pragma unroll
  for (int j = 0; j < 8; ++j)
#pragma unroll
    for (int r = 0; r < 8; ++r) {
      const int m = r + (hi ? 8 : 0);
      const size_t o = ((size_t)b * TT + t0 + m) * DD + c0 + j * 16 + l16;
      pspv[o] = (_Float16)(acc[j][r] + v_fea[o]);
    }
}

// 8) pspa = a_fea + g_av @ v1; A tile = masked/scaled beta columns (strided),
//    built once per k-step and reused for 8 WMMAs.
__global__ void psp_attn_av(const float* __restrict__ beta, const float* __restrict__ thrC,
                            const float* __restrict__ scaleC, const _Float16* __restrict__ v1T,
                            const float* __restrict__ a_fea, _Float16* __restrict__ pspa) {
  const int tid = threadIdx.x, wid = tid >> 5, lane = tid & 31;
  const int l16 = lane & 15, hi = lane >> 4;
  const int w = blockIdx.x * 8 + wid;             // 2048 waves
  const int c0 = (w & 1) * 128;
  const int s0 = ((w >> 1) & 127) * 16;
  const int b  = w >> 8;
  const int scol = b * TT + s0 + l16;
  const float tc = thrC[scol], sc = scaleC[scol];
  const float* bcol = beta + (size_t)b * TT * TT + (s0 + l16);
  const _Float16* bTb = v1T + ((size_t)b * DD + c0) * TT;
  v8f acc[8] = {};
  for (int ti = 0; ti < 64; ++ti) {
    const int t0 = ti * 32;
    const int tb = t0 + (hi ? 16 : 0);
    __builtin_prefetch(bcol + (size_t)(tb + 64) * TT, 0, 3);
    v16h a;
#pragma unroll
    for (int j = 0; j < 8; ++j) {
      float v0 = bcol[(size_t)(tb + j) * TT];
      float v1 = bcol[(size_t)(tb + 8 + j) * TT];
      a[j]     = (_Float16)(v0 > tc ? v0 * sc : 0.f);
      a[8 + j] = (_Float16)(v1 > tc ? v1 * sc : 0.f);
    }
#pragma unroll
    for (int j = 0; j < 8; ++j) {
      v16h bv = *(const v16h*)(bTb + (size_t)(j * 16 + l16) * TT + t0 + (hi ? 16 : 0));
      acc[j] = wmma16(a, bv, acc[j]);
    }
  }
#pragma unroll
  for (int j = 0; j < 8; ++j)
#pragma unroll
    for (int r = 0; r < 8; ++r) {
      const int m = r + (hi ? 8 : 0);
      const size_t o = ((size_t)b * TT + s0 + m) * DD + c0 + j * 16 + l16;
      pspa[o] = (_Float16)(acc[j][r] + a_fea[o]);
    }
}

// ---------------------------------------------------------------------------
// 9) out = LayerNorm(relu(psp @ Wh^T)). Wave pairs cooperate: 8 waves/block =
//    4 row-groups x 2 column-halves; LN stats exchanged through LDS.
__global__ void psp_fc_ln(const _Float16* __restrict__ psp, const _Float16* __restrict__ Wh,
                          const float* __restrict__ lng, const float* __restrict__ lnb,
                          float* __restrict__ out) {
  __shared__ float lds_s1[4][16][2];
  __shared__ float lds_s2[4][16][2];
  const int tid = threadIdx.x, wid = tid >> 5, lane = tid & 31;
  const int l16 = lane & 15, hi = lane >> 4;
  const int rg = wid >> 1;                        // row-group within block (0..3)
  const int half = wid & 1;                       // column half (0..1)
  const int c0 = half * 128;
  const int r0 = (blockIdx.x * 4 + rg) * 16;      // 1024 row tiles
  const _Float16* ar = psp + (size_t)(r0 + l16) * DD;
  v8f acc[8] = {};
#pragma unroll
  for (int kk = 0; kk < 8; ++kk) {
    const int k0 = kk * 32;
    v16h a = loadA_h(ar + k0 + (hi ? 16 : 0));
#pragma unroll
    for (int j = 0; j < 8; ++j) {
      v16h bv = *(const v16h*)(Wh + (size_t)(c0 + j * 16 + l16) * DD + k0 + (hi ? 16 : 0));
      acc[j] = wmma16(a, bv, acc[j]);
    }
  }
  // relu + per-row partial sums over this wave's 128 columns
#pragma unroll
  for (int r = 0; r < 8; ++r) {
    float s1 = 0.f, s2 = 0.f;
#pragma unroll
    for (int j = 0; j < 8; ++j) {
      float v = acc[j][r]; v = v > 0.f ? v : 0.f;
      acc[j][r] = v; s1 += v; s2 += v * v;
    }
    // reduce within 16-lane halves (rows r and r+8 simultaneously)
#pragma unroll
    for (int m = 1; m < 16; m <<= 1) { s1 += __shfl_xor(s1, m, 32); s2 += __shfl_xor(s2, m, 32); }
    if (l16 == 0) {                               // lane 0 (row r) and lane 16 (row r+8)
      const int row = r + (hi ? 8 : 0);
      lds_s1[rg][row][half] = s1;
      lds_s2[rg][row][half] = s2;
    }
  }
  __syncthreads();
#pragma unroll
  for (int r = 0; r < 8; ++r) {
    const int row = r + (hi ? 8 : 0);
    const float t1 = lds_s1[rg][row][0] + lds_s1[rg][row][1];
    const float t2 = lds_s2[rg][row][0] + lds_s2[rg][row][1];
    const float mean = t1 * (1.0f / 256.0f);
    const float var  = t2 * (1.0f / 256.0f) - mean * mean;
    const float rs = rsqrtf(var + 1e-6f);
    float* orow = out + (size_t)(r0 + row) * DD;
#pragma unroll
    for (int j = 0; j < 8; ++j) {
      const int col = c0 + j * 16 + l16;
      orow[col] = (acc[j][r] - mean) * rs * lng[col] + lnb[col];
    }
  }
}

// 10) fuse
__global__ void psp_fuse(const float* __restrict__ v, const float* __restrict__ a,
                         float* __restrict__ o, int n) {
  const int i = blockIdx.x * blockDim.x + threadIdx.x;
  if (i < n) o[i] = 0.5f * (v[i] + a[i]);
}

// ---------------------------------------------------------------------------
extern "C" void kernel_launch(void* const* d_in, const int* in_sizes, int n_in,
                              void* d_out, int out_size, void* d_ws, size_t ws_size,
                              hipStream_t stream) {
  (void)in_sizes; (void)n_in; (void)out_size; (void)ws_size;
  const float* a_fea = (const float*)d_in[0];
  const float* v_fea = (const float*)d_in[1];
  const float* thr   = (const float*)d_in[2];
  const float* Wv1   = (const float*)d_in[3];
  const float* Wv2   = (const float*)d_in[4];
  const float* Wvfc  = (const float*)d_in[5];
  const float* Wa1   = (const float*)d_in[6];
  const float* Wa2   = (const float*)d_in[7];
  const float* Wafc  = (const float*)d_in[8];
  const float* ln_g  = (const float*)d_in[9];
  const float* ln_b  = (const float*)d_in[10];
  float* out = (float*)d_out;

  const size_t BT = (size_t)BB * TT;              // 16384
  char* ws = (char*)d_ws;
  // workspace layout (~161 MB; beta kept f32 to match reference numerics)
  float*    beta = (float*)ws;                                    // 134217728 B
  _Float16* v2h  = (_Float16*)(ws + 134217728);                   //   8388608 B
  _Float16* a1h  = (_Float16*)(ws + 142606336);                   //   8388608 B
  _Float16* v1T  = (_Float16*)(ws + 150994944);                   //   8388608 B
  _Float16* a2T  = (_Float16*)(ws + 159383552);                   //   8388608 B
  float*    S    = (float*)(ws + 167772160);                      // 6 x 65536 B
  float*    thrR = S + BT;
  float*    scR  = thrR + BT;
  float*    C    = scR + BT;
  float*    thrC = C + BT;
  float*    scC  = thrC + BT;
  _Float16* Wh   = (_Float16*)(ws + 168165376);                   // 6 x 131072 B
  _Float16* Wv1h = Wh,            *Wv2h = Wh + 65536,   *Wa1h = Wh + 131072;
  _Float16* Wa2h = Wh + 196608,   *Wvfch = Wh + 262144, *Wafch = Wh + 327680;
  _Float16* pspv = v2h;   // reuse: v2h/a1h dead after psp_beta_row
  _Float16* pspa = a1h;

  const dim3 blk(256);
  psp_cvt_f16<<<256, blk, 0, stream>>>(Wv1, Wv1h, 65536);
  psp_cvt_f16<<<256, blk, 0, stream>>>(Wv2, Wv2h, 65536);
  psp_cvt_f16<<<256, blk, 0, stream>>>(Wa1, Wa1h, 65536);
  psp_cvt_f16<<<256, blk, 0, stream>>>(Wa2, Wa2h, 65536);
  psp_cvt_f16<<<256, blk, 0, stream>>>(Wvfc, Wvfch, 65536);
  psp_cvt_f16<<<256, blk, 0, stream>>>(Wafc, Wafch, 65536);

  psp_proj_row<<<256, blk, 0, stream>>>(v_fea, Wv1h, nullptr, v1T);
  psp_proj_row<<<256, blk, 0, stream>>>(v_fea, Wv2h, v2h, nullptr);
  psp_proj_row<<<256, blk, 0, stream>>>(a_fea, Wa1h, a1h, nullptr);
  psp_proj_row<<<256, blk, 0, stream>>>(a_fea, Wa2h, nullptr, a2T);

  psp_beta_row<<<2048, blk, 0, stream>>>(v2h, a1h, beta);

  psp_rowsum<<<2048, blk, 0, stream>>>(beta, S);
  psp_colsum<<<64, blk, 0, stream>>>(beta, C);
  psp_rowmask<<<2048, blk, 0, stream>>>(beta, S, thr, thrR, scR);
  psp_colmask<<<64, blk, 0, stream>>>(beta, C, thr, thrC, scC, out + 3 * BT * DD);

  psp_attn_va<<<256, blk, 0, stream>>>(beta, thrR, scR, a2T, v_fea, pspv);
  psp_attn_av<<<256, blk, 0, stream>>>(beta, thrC, scC, v1T, a_fea, pspa);

  psp_fc_ln<<<256, blk, 0, stream>>>(pspv, Wvfch, ln_g, ln_b, out + BT * DD);
  psp_fc_ln<<<256, blk, 0, stream>>>(pspa, Wafch, ln_g, ln_b, out + 2 * BT * DD);
  psp_fuse<<<16384, blk, 0, stream>>>(out + BT * DD, out + 2 * BT * DD, out, (int)(BT * DD));
}
